// MaxSimRanker_70789650972852
// MI455X (gfx1250) — compile-verified, matrix operand order checked by reading, not run on hardware
//
#include <hip/hip_runtime.h>
#include <hip/hip_bf16.h>
#include <math.h>

#define NDOCS  5000
#define DOCLEN 128
#define DIM    128
#define BB     8
#define NQ     32
#define NTOK   1024
#define TOPK   100

typedef __attribute__((ext_vector_type(16))) __bf16 v16bf;
typedef __attribute__((ext_vector_type(8)))  float  v8f;

// ---------------------------------------------------------------------------
// Kernel 0: convert Q to bf16 once, pre-swizzled into WMMA per-lane order.
// Chunk index = ((b*2 + m)*4 + kk)*32 + lane ; each chunk = 16 bf16 (32 B).
// Lane layout (16-bit A 16x32): half = lane>>4, row = m*16 + (lane&15),
// elems 0-7 = dims kk*32 + half*8 + [0..7], elems 8-15 = +16 further.
// ---------------------------------------------------------------------------
__global__ __launch_bounds__(256) void pack_q_bf16(
    const float* __restrict__ q, __bf16* __restrict__ qbf) {
  const int idx = blockIdx.x * 256 + threadIdx.x;     // 8*2*4*32 = 2048 chunks
  if (idx >= BB * 2 * 4 * 32) return;
  const int lane = idx & 31;
  const int kk   = (idx >> 5) & 3;
  const int m    = (idx >> 7) & 1;
  const int b    = idx >> 8;
  const int half = lane >> 4;
  const int row  = m * 16 + (lane & 15);
  const float* src = q + ((size_t)(b * NQ + row)) * DIM + kk * 32 + half * 8;
  __bf16* dst = qbf + (size_t)idx * 16;
#pragma unroll
  for (int e = 0; e < 8; ++e) {
    dst[e]     = (__bf16)src[e];
    dst[8 + e] = (__bf16)src[16 + e];
  }
}

// ---------------------------------------------------------------------------
// Kernel 1: per-batch unique candidate pids (deterministic, ascending order)
// ---------------------------------------------------------------------------
__global__ __launch_bounds__(256) void build_candidates(
    const int* __restrict__ token_ids, const int* __restrict__ emb2pid,
    int* __restrict__ cand_pids, float* __restrict__ cand_scores,
    int* __restrict__ cand_count) {
  __shared__ unsigned char flag[NDOCS];
  __shared__ int counts[256];
  const int b = blockIdx.x;
  const int tid = threadIdx.x;

  for (int i = tid; i < NDOCS; i += 256) flag[i] = 0;
  __syncthreads();
  for (int t = tid; t < NTOK; t += 256) {
    int tok = token_ids[b * NTOK + t];
    int pid = emb2pid[tok];
    if (pid >= 0 && pid < NDOCS) flag[pid] = 1;   // races benign: all write 1
  }
  __syncthreads();

  const int CH = (NDOCS + 255) / 256;             // 20 pids per thread
  const int lo = tid * CH;
  const int hi = min(lo + CH, NDOCS);
  int my = 0;
  for (int p = lo; p < hi; ++p) my += flag[p];
  counts[tid] = my;
  __syncthreads();
  // Hillis-Steele inclusive scan
  for (int off = 1; off < 256; off <<= 1) {
    int v = (tid >= off) ? counts[tid - off] : 0;
    __syncthreads();
    counts[tid] += v;
    __syncthreads();
  }
  int ofs = counts[tid] - my;
  const int total = counts[255];
  for (int p = lo; p < hi; ++p)
    if (flag[p]) cand_pids[b * NTOK + (ofs++)] = p;
  for (int i = total + tid; i < NTOK; i += 256) cand_pids[b * NTOK + i] = -1;
  for (int i = tid; i < NTOK; i += 256) cand_scores[b * NTOK + i] = -INFINITY;
  if (tid == 0) cand_count[b] = total;
}

// ---------------------------------------------------------------------------
// Kernel 2: MaxSim scoring with v_wmma_f32_16x16x32_bf16
// One block per (batch, candidate); wave w owns doc-token columns [16w,16w+16)
// A (queries) comes pre-converted/pre-swizzled bf16; B (doc rows) converted
// on the fly from fp32.
// ---------------------------------------------------------------------------
__device__ __forceinline__ v16bf load16f(const float* __restrict__ p) {
  // Per-lane B layout: 8 contiguous K at p, 8 more at p+16 (ISA 16-bit 16x32)
  const float4 a = *(const float4*)(p);
  const float4 b = *(const float4*)(p + 4);
  const float4 c = *(const float4*)(p + 16);
  const float4 d = *(const float4*)(p + 20);
  v16bf r;
  r[0]  = (__bf16)a.x; r[1]  = (__bf16)a.y; r[2]  = (__bf16)a.z; r[3]  = (__bf16)a.w;
  r[4]  = (__bf16)b.x; r[5]  = (__bf16)b.y; r[6]  = (__bf16)b.z; r[7]  = (__bf16)b.w;
  r[8]  = (__bf16)c.x; r[9]  = (__bf16)c.y; r[10] = (__bf16)c.z; r[11] = (__bf16)c.w;
  r[12] = (__bf16)d.x; r[13] = (__bf16)d.y; r[14] = (__bf16)d.z; r[15] = (__bf16)d.w;
  return r;
}

__global__ __launch_bounds__(256) void maxsim_score(
    const __bf16* __restrict__ qbf, const float* __restrict__ vecs,
    const int* __restrict__ cand_pids, float* __restrict__ cand_scores) {
  const int b = blockIdx.y;
  const int c = blockIdx.x;
  const int pid = cand_pids[b * NTOK + c];
  if (pid < 0) return;

  const int tid  = threadIdx.x;
  const int wave = tid >> 5;        // 0..7 -> doc-token column tile
  const int lane = tid & 31;
  const int half = lane >> 4;       // K-split half per ISA layout
  const int row  = lane & 15;

  __shared__ float rowmax[8][NQ];   // per-wave, per-q max over 16 doc columns

  const float* db   = vecs + (size_t)pid * DOCLEN * DIM;
  const float* brow = db + (size_t)(wave * 16 + row) * DIM + half * 8;
  // A chunks: ((b*2 + m)*4 + kk)*32 + lane, 16 bf16 each
  const v16bf* a0 = (const v16bf*)(qbf + ((size_t)(b * 2 + 0) * 4 * 32 + lane) * 16);
  const v16bf* a1 = (const v16bf*)(qbf + ((size_t)(b * 2 + 1) * 4 * 32 + lane) * 16);

  v8f acc0 = {};  // q rows 0-15  x doc cols
  v8f acc1 = {};  // q rows 16-31 x doc cols
#pragma unroll
  for (int kk = 0; kk < DIM / 32; ++kk) {
    const int kb = kk * 32;
    if (kk + 1 < DIM / 32) __builtin_prefetch(brow + kb + 32, 0, 3);
    v16bf A0 = a0[kk * 32];         // stride between kk chunks = 32 lanes
    v16bf A1 = a1[kk * 32];
    v16bf Bm = load16f(brow + kb);
    acc0 = __builtin_amdgcn_wmma_f32_16x16x32_bf16(false, A0, false, Bm,
                                                   (short)0, acc0, false, false);
    acc1 = __builtin_amdgcn_wmma_f32_16x16x32_bf16(false, A1, false, Bm,
                                                   (short)0, acc1, false, false);
  }

  // Max over the 16 doc columns living across each 16-lane half
#pragma unroll
  for (int m = 1; m <= 8; m <<= 1) {
#pragma unroll
    for (int e = 0; e < 8; ++e) {
      acc0[e] = fmaxf(acc0[e], __shfl_xor(acc0[e], m, 16));
      acc1[e] = fmaxf(acc1[e], __shfl_xor(acc1[e], m, 16));
    }
  }
  // C layout: VGPR e = q-row e (lanes 0-15) / q-row 8+e (lanes 16-31)
  if (row == 0) {
#pragma unroll
    for (int e = 0; e < 8; ++e) {
      rowmax[wave][half * 8 + e]      = acc0[e];
      rowmax[wave][16 + half * 8 + e] = acc1[e];
    }
  }
  __syncthreads();

  if (tid < NQ) {
    float mx = rowmax[0][tid];
#pragma unroll
    for (int w = 1; w < 8; ++w) mx = fmaxf(mx, rowmax[w][tid]);
#pragma unroll
    for (int m = 1; m < 32; m <<= 1) mx += __shfl_xor(mx, m, 32);
    if (tid == 0) cand_scores[b * NTOK + c] = mx;
  }
}

// ---------------------------------------------------------------------------
// Kernel 3: top-100 per batch (iterative argmax, lowest-index tie-break)
// ---------------------------------------------------------------------------
__global__ __launch_bounds__(256) void topk_select(
    const float* __restrict__ cand_scores, const int* __restrict__ cand_pids,
    float* __restrict__ out) {
  const int b = blockIdx.x;
  const int tid = threadIdx.x;
  __shared__ float vals[NTOK];
  __shared__ int   ip[NTOK];
  __shared__ unsigned char used[NTOK];
  __shared__ float bv[256];
  __shared__ int   bi[256];

  for (int i = tid; i < NTOK; i += 256) {
    vals[i] = cand_scores[b * NTOK + i];
    ip[i]   = cand_pids[b * NTOK + i];
    used[i] = 0;
  }
  __syncthreads();

  for (int k = 0; k < TOPK; ++k) {
    float best = -INFINITY;
    int besti = NTOK;
    for (int i = tid; i < NTOK; i += 256) {
      if (!used[i]) {
        float v = vals[i];
        if (v > best || (v == best && i < besti)) { best = v; besti = i; }
      }
    }
    bv[tid] = best; bi[tid] = besti;
    __syncthreads();
    for (int off = 128; off > 0; off >>= 1) {
      if (tid < off) {
        float ov = bv[tid + off]; int oi = bi[tid + off];
        if (ov > bv[tid] || (ov == bv[tid] && oi < bi[tid])) { bv[tid] = ov; bi[tid] = oi; }
      }
      __syncthreads();
    }
    if (tid == 0) {
      int j = bi[0];
      if (j >= NTOK) j = NTOK - 1;
      out[b * TOPK + k]             = bv[0];
      out[BB * TOPK + b * TOPK + k] = (float)ip[j];
      used[j] = 1;
    }
    __syncthreads();
  }
}

// ---------------------------------------------------------------------------
extern "C" void kernel_launch(void* const* d_in, const int* in_sizes, int n_in,
                              void* d_out, int out_size, void* d_ws, size_t ws_size,
                              hipStream_t stream) {
  (void)in_sizes; (void)n_in; (void)out_size; (void)ws_size;
  const float* q         = (const float*)d_in[0];  // [8,32,128] f32
  const int*   token_ids = (const int*)  d_in[1];  // [8,1024]
  const float* vecs      = (const float*)d_in[2];  // [5000,128,128] f32
  const int*   emb2pid   = (const int*)  d_in[3];  // [640000]
  // d_in[4] = k (==100, fixed as TOPK to match out_size)

  int*    cand_pids   = (int*)d_ws;
  float*  cand_scores = (float*)((char*)d_ws + (size_t)BB * NTOK * sizeof(int));
  int*    cand_count  = (int*)((char*)d_ws + (size_t)2 * BB * NTOK * 4);
  __bf16* qbf         = (__bf16*)((char*)d_ws + 66048);  // 256B-aligned, 64KB

  pack_q_bf16<<<BB, 256, 0, stream>>>(q, qbf);
  build_candidates<<<BB, 256, 0, stream>>>(token_ids, emb2pid, cand_pids,
                                           cand_scores, cand_count);
  dim3 g2(NTOK, BB);
  maxsim_score<<<g2, 256, 0, stream>>>(qbf, vecs, cand_pids, cand_scores);
  topk_select<<<BB, 256, 0, stream>>>(cand_scores, cand_pids, (float*)d_out);
}
